// _MultiHeadScaledDotAttention_34144990003366
// MI455X (gfx1250) — matrix operand order repaired and checked by reading, not compiled, not run on hardware
//
#include <hip/hip_runtime.h>

typedef _Float16 half_t;
typedef __attribute__((ext_vector_type(16))) _Float16 v16h;
typedef __attribute__((ext_vector_type(8)))  _Float16 v8h;
typedef __attribute__((ext_vector_type(8)))  float    v8f;
typedef __attribute__((ext_vector_type(4)))  unsigned int v4u;
typedef __attribute__((ext_vector_type(8)))  int      v8i;
typedef __attribute__((ext_vector_type(4)))  int      v4i;

#define B_   4
#define T_   2048
#define H_   16
#define U_   64
#define D_   1024
#define SCALE 0.125f        // 1/sqrt(64)
#define LBIAS 10000.0f
#define NKT  (T_ / 64)      // key tiles of 64

static __device__ __forceinline__ v8f wmma16(v16h a, v16h b, v8f c) {
    return __builtin_amdgcn_wmma_f32_16x16x32_f16(false, a, false, b,
                                                  (short)0, c, false, false);
}

static __device__ __forceinline__ v16h ld_contig16(const half_t* p) {
    v8h lo = *(const v8h*)p;
    v8h hh = *(const v8h*)(p + 8);
    v16h f;
#pragma unroll
    for (int e = 0; e < 8; ++e) { f[e] = lo[e]; f[e + 8] = hh[e]; }
    return f;
}

static __device__ __forceinline__ v16h ld_split16(const half_t* p0, const half_t* p1) {
    v8h lo = *(const v8h*)p0;
    v8h hh = *(const v8h*)p1;
    v16h f;
#pragma unroll
    for (int e = 0; e < 8; ++e) { f[e] = lo[e]; f[e + 8] = hh[e]; }
    return f;
}

// 16-byte async copy global -> LDS (ASYNCcnt-tracked, per-lane addresses)
static __device__ __forceinline__ void async_copy16(uint32_t lds_off, const void* gptr) {
    asm volatile("global_load_async_to_lds_b128 %0, %1, off"
                 :: "v"(lds_off), "v"(gptr) : "memory");
}
static __device__ __forceinline__ void wait_async_le4() {
    asm volatile("s_wait_asynccnt 4" ::: "memory");
}
static __device__ __forceinline__ void wait_async_le0() {
    asm volatile("s_wait_asynccnt 0" ::: "memory");
}

// ---------------------------------------------------------------------------
// TDM: 2-D f16 tile load (64 rows x 64 elements), padded to 72-half LDS rows.
//   pad_interval = 4 -> pad after every 32 DWORDs (one 128B row)
//   pad_amount   = 3 -> insert 4 DWORDs (16B) of padding
// ---------------------------------------------------------------------------
static __device__ __forceinline__ void tdm_load_2d(uint32_t lds_addr, const void* gaddr,
                                                   uint32_t tensor_d0, uint32_t tensor_d1,
                                                   uint32_t stride0) {
    unsigned long long ga = (unsigned long long)(size_t)gaddr;
    v4u g0;
    g0.x = 1u;                                            // count=1 (valid), user mode
    g0.y = lds_addr;                                      // LDS byte address
    g0.z = (unsigned)(ga & 0xFFFFFFFFu);                  // global_addr[31:0]
    g0.w = (unsigned)((ga >> 32) & 0x01FFFFFFu) | (2u << 30);   // addr[56:32] | type=2
    v8i g1;
    g1[0] = (int)((1u << 16) | (1u << 20) | (4u << 22) | (3u << 25));
            // workgroup_mask=0, data_size=1 (2B), pad_enable, interval=32DW, amount=4DW
    g1[1] = (int)((tensor_d0 & 0xFFFFu) << 16);           // abar=0 | tensor_dim0 lo16
    g1[2] = (int)(((tensor_d0 >> 16) & 0xFFFFu) | ((tensor_d1 & 0xFFFFu) << 16));
    g1[3] = (int)(((tensor_d1 >> 16) & 0xFFFFu) | (64u << 16));  // tile_dim0 = 64
    g1[4] = (int)64u;                                     // tile_dim1 = 64, tile_dim2 = 0
    g1[5] = (int)stride0;                                 // tensor_dim0_stride lo32
    g1[6] = 0;                                            // stride0 hi16 | dim1_stride lo16
    g1[7] = 0;
    v4i g2 = {0, 0, 0, 0};
    v4i g3 = {0, 0, 0, 0};
    v8i g4 = {0, 0, 0, 0, 0, 0, 0, 0};
    __builtin_amdgcn_tensor_load_to_lds(g0, g1, g2, g3, g4, 0);
}

// ---------------------------------------------------------------------------
// One-time conversions into workspace
// ---------------------------------------------------------------------------
__global__ __launch_bounds__(256)
void wconv_kernel(const float* __restrict__ W, half_t* __restrict__ Wt) {
    int idx = blockIdx.x * 256 + threadIdx.x;              // 1M
    int n = idx & (D_ - 1);
    int kr = idx >> 10;
    Wt[(size_t)n * D_ + kr] = (half_t)W[(size_t)kr * D_ + n];
}

__global__ __launch_bounds__(256)
void kconv_kernel(const float* __restrict__ k, half_t* __restrict__ Kh) {
    size_t idx = (size_t)blockIdx.x * 256 + threadIdx.x;   // 8M, same layout
    Kh[idx] = (half_t)k[idx];
}

// V fp32 [b][t][h*U+u] -> f16 Vt [b][h][u][t]  (write-contiguous)
__global__ __launch_bounds__(256)
void vconv_kernel(const float* __restrict__ v, half_t* __restrict__ Vt) {
    size_t idx = (size_t)blockIdx.x * 256 + threadIdx.x;   // 8M
    int    t  = (int)(idx & (T_ - 1));
    size_t r  = idx >> 11;            // (b*H + h)*U + u
    int    u  = (int)(r & (U_ - 1));
    size_t bh = r >> 6;
    int    h  = (int)(bh & (H_ - 1));
    size_t b  = bh >> 4;
    Vt[idx] = (half_t)v[(b * T_ + t) * D_ + (size_t)h * U_ + u];
}

// ---------------------------------------------------------------------------
// Flash attention: 8 wave32s per block, 128 Q rows, TDM double-buffered K/V
// ---------------------------------------------------------------------------
__global__ __launch_bounds__(256)
void attn_kernel(const float* __restrict__ q, const half_t* __restrict__ Kh,
                 const half_t* __restrict__ Vt, const int* __restrict__ mask,
                 half_t* __restrict__ att) {
    __shared__ half_t sK[2][64][72];     // [buf][key][u]
    __shared__ half_t sV[2][64][72];     // [buf][u][key]
    __shared__ half_t sP[8][16][72];     // per-wave P tile

    const int bid  = blockIdx.x;
    const int qt   = bid & 15;           // T_/128
    const int h    = (bid >> 4) & 15;
    const int b    = bid >> 8;
    const int tid  = threadIdx.x;
    const int lane = tid & 31;
    const int wave = tid >> 5;
    const int lcol = lane & 15;
    const int hi   = lane >> 4;

    // ---- wave 0 drives the Tensor Data Mover (2 descriptors per tile) ----
    auto issue_tdm = [&](int kt, int buf) {
        const half_t* gk = Kh + ((size_t)(b * T_ + kt * 64)) * D_ + h * U_;
        tdm_load_2d((uint32_t)(size_t)&sK[buf][0][0], gk, 64, 64, D_);
        const half_t* gv = Vt + ((size_t)(b * H_ + h) * U_) * T_ + kt * 64;
        tdm_load_2d((uint32_t)(size_t)&sV[buf][0][0], gv, T_, 64, T_);
    };

    // ---- Q A-fragments resident in registers ----
    v16h aQ[2];
    {
        const float* qb = q + ((size_t)b * T_ + (qt * 128 + wave * 16 + lcol)) * D_ + h * U_;
#pragma unroll
        for (int c = 0; c < 2; ++c) {
            const float* s0 = qb + c * 32 + hi * 8;
            const float* s1 = qb + c * 32 + 16 + hi * 8;
            float4 a0 = *(const float4*)(s0);
            float4 a1 = *(const float4*)(s0 + 4);
            float4 b0 = *(const float4*)(s1);
            float4 b1 = *(const float4*)(s1 + 4);
            v16h f;
            f[0]=(half_t)a0.x; f[1]=(half_t)a0.y; f[2]=(half_t)a0.z; f[3]=(half_t)a0.w;
            f[4]=(half_t)a1.x; f[5]=(half_t)a1.y; f[6]=(half_t)a1.z; f[7]=(half_t)a1.w;
            f[8]=(half_t)b0.x; f[9]=(half_t)b0.y; f[10]=(half_t)b0.z; f[11]=(half_t)b0.w;
            f[12]=(half_t)b1.x; f[13]=(half_t)b1.y; f[14]=(half_t)b1.z; f[15]=(half_t)b1.w;
            aQ[c] = f;
        }
    }

    v8f o[4];
#pragma unroll
    for (int n = 0; n < 4; ++n) o[n] = (v8f){0.f,0.f,0.f,0.f,0.f,0.f,0.f,0.f};
    float mrow[8], lrow[8];
#pragma unroll
    for (int i = 0; i < 8; ++i) { mrow[i] = -3.0e38f; lrow[i] = 0.0f; }

    if (wave == 0) issue_tdm(0, 0);

    for (int kt = 0; kt < NKT; ++kt) {
        const int cur = kt & 1;
        if (wave == 0) {
            if (kt + 1 < NKT) {
                issue_tdm(kt + 1, cur ^ 1);
                __builtin_amdgcn_s_wait_tensorcnt(2);    // current tile's 2 TDMs done
            } else {
                __builtin_amdgcn_s_wait_tensorcnt(0);
            }
        }
        __syncthreads();                                  // publish tile to all waves

        // ---- S = Q K^T ----
        v8f s[4];
#pragma unroll
        for (int n = 0; n < 4; ++n) {
            v8f acc = (v8f){0.f,0.f,0.f,0.f,0.f,0.f,0.f,0.f};
#pragma unroll
            for (int c = 0; c < 2; ++c) {
                v16h bf = ld_split16(&sK[cur][n * 16 + lcol][c * 32 + hi * 8], &sK[cur][n * 16 + lcol][c * 32 + 16 + hi * 8]);
                acc = wmma16(aQ[c], bf, acc);
            }
            s[n] = acc;
        }

        // ---- scale + additive mask bias ----
#pragma unroll
        for (int n = 0; n < 4; ++n) {
            int keyg = kt * 64 + n * 16 + lcol;
            float bias = (1.0f - (float)mask[b * T_ + keyg]) * LBIAS;
#pragma unroll
            for (int i = 0; i < 8; ++i) s[n][i] = s[n][i] * SCALE - bias;
        }

        // ---- online softmax (row M = i + 8*hi) ----
        float alpha[8];
#pragma unroll
        for (int i = 0; i < 8; ++i) {
            float mx = fmaxf(fmaxf(s[0][i], s[1][i]), fmaxf(s[2][i], s[3][i]));
#pragma unroll
            for (int off = 1; off < 16; off <<= 1)
                mx = fmaxf(mx, __shfl_xor(mx, off, 32));
            float mn = fmaxf(mrow[i], mx);
            alpha[i] = __expf(mrow[i] - mn);
            mrow[i] = mn;
            float rs = 0.0f;
#pragma unroll
            for (int n = 0; n < 4; ++n) {
                float p = __expf(s[n][i] - mn);
                s[n][i] = p;
                rs += p;
            }
#pragma unroll
            for (int off = 1; off < 16; off <<= 1)
                rs += __shfl_xor(rs, off, 32);
            lrow[i] = lrow[i] * alpha[i] + rs;
        }

        // ---- P (C-layout) -> LDS -> A-layout ----
#pragma unroll
        for (int n = 0; n < 4; ++n)
#pragma unroll
            for (int i = 0; i < 8; ++i)
                sP[wave][i + 8 * hi][n * 16 + lcol] = (half_t)s[n][i];
        __syncthreads();

#pragma unroll
        for (int nu = 0; nu < 4; ++nu)
#pragma unroll
            for (int i = 0; i < 8; ++i)
                o[nu][i] *= alpha[i];

        v16h aP[2];
#pragma unroll
        for (int c = 0; c < 2; ++c) {
            const half_t* pr = &sP[wave][lcol][0];
            aP[c] = ld_split16(pr + c * 32 + hi * 8, pr + c * 32 + 16 + hi * 8);
        }
#pragma unroll
        for (int nu = 0; nu < 4; ++nu) {
#pragma unroll
            for (int c = 0; c < 2; ++c) {
                v16h bf = ld_split16(&sV[cur][nu * 16 + lcol][c * 32 + hi * 8], &sV[cur][nu * 16 + lcol][c * 32 + 16 + hi * 8]);
                o[nu] = wmma16(aP[c], bf, o[nu]);
            }
        }
        __syncthreads();
    }

    // ---- normalize and store att (f16) ----
    float rcp[8];
#pragma unroll
    for (int i = 0; i < 8; ++i) rcp[i] = 1.0f / lrow[i];
    half_t* ab = att + ((size_t)b * T_ + (qt * 128 + wave * 16)) * D_ + h * U_;
#pragma unroll
    for (int n = 0; n < 4; ++n)
#pragma unroll
        for (int i = 0; i < 8; ++i)
            ab[(size_t)(i + 8 * hi) * D_ + n * 16 + lcol] = (half_t)(o[n][i] * rcp[i]);
}

// ---------------------------------------------------------------------------
// Projection GEMM: 128x128 tiles, async double-buffered A/B staging
// ---------------------------------------------------------------------------
__global__ __launch_bounds__(256)
void proj_kernel(const half_t* __restrict__ A, const half_t* __restrict__ Bt,
                 float* __restrict__ out) {
    __shared__ half_t sA[2][128][40];    // [buf][m][k]
    __shared__ half_t sB[2][128][40];    // [buf][n][k]

    const int bid  = blockIdx.x;
    const int nb   = bid & 7;
    const int mb   = bid >> 3;
    const int m0   = mb * 128;
    const int n0   = nb * 128;
    const int tid  = threadIdx.x;
    const int lane = tid & 31;
    const int wave = tid >> 5;
    const int lcol = lane & 15;
    const int hi   = lane >> 4;
    const int wm   = wave & 3;
    const int wn   = wave >> 2;

    auto issue_chunk = [&](int kc, int buf) {
#pragma unroll
        for (int i = 0; i < 2; ++i) {
            int chunk = tid + i * 256;               // 512 chunks per array
            int row = chunk >> 2;                    // 4 x 16B segs per 64B row
            int seg = chunk & 3;
            uint32_t da = (uint32_t)(size_t)&sA[buf][row][seg * 8];
            const half_t* ga = A + (size_t)(m0 + row) * D_ + kc * 32 + seg * 8;
            async_copy16(da, ga);
            uint32_t db = (uint32_t)(size_t)&sB[buf][row][seg * 8];
            const half_t* gb = Bt + (size_t)(n0 + row) * D_ + kc * 32 + seg * 8;
            async_copy16(db, gb);
        }
    };

    v8f acc[2][4];
#pragma unroll
    for (int mi = 0; mi < 2; ++mi)
#pragma unroll
        for (int ni = 0; ni < 4; ++ni)
            acc[mi][ni] = (v8f){0.f,0.f,0.f,0.f,0.f,0.f,0.f,0.f};

    issue_chunk(0, 0);

    for (int kc = 0; kc < D_ / 32; ++kc) {
        const int cur = kc & 1;
        if (kc + 1 < D_ / 32) { issue_chunk(kc + 1, cur ^ 1); wait_async_le4(); }
        else                  { wait_async_le0(); }
        __syncthreads();

        v16h aF[2], bF[4];
#pragma unroll
        for (int mi = 0; mi < 2; ++mi) {
            const half_t* pr = &sA[cur][wm * 32 + mi * 16 + lcol][0];
            aF[mi] = ld_split16(pr + hi * 8, pr + 16 + hi * 8);
        }
#pragma unroll
        for (int ni = 0; ni < 4; ++ni)
            bF[ni] = ld_split16(&sB[cur][wn * 64 + ni * 16 + lcol][hi * 8], &sB[cur][wn * 64 + ni * 16 + lcol][16 + hi * 8]);

#pragma unroll
        for (int mi = 0; mi < 2; ++mi)
#pragma unroll
            for (int ni = 0; ni < 4; ++ni)
                acc[mi][ni] = wmma16(aF[mi], bF[ni], acc[mi][ni]);
        __syncthreads();
    }

#pragma unroll
    for (int mi = 0; mi < 2; ++mi)
#pragma unroll
        for (int ni = 0; ni < 4; ++ni) {
            float* ob = out + (size_t)(m0 + wm * 32 + mi * 16 + 8 * hi) * D_
                            + n0 + wn * 64 + ni * 16 + lcol;
#pragma unroll
            for (int i = 0; i < 8; ++i)
                ob[(size_t)i * D_] = acc[mi][ni][i];
        }
}

// ---------------------------------------------------------------------------
extern "C" void kernel_launch(void* const* d_in, const int* in_sizes, int n_in,
                              void* d_out, int out_size, void* d_ws, size_t ws_size,
                              hipStream_t stream) {
    const float* q    = (const float*)d_in[0];
    const float* kmat = (const float*)d_in[1];
    const float* vmat = (const float*)d_in[2];
    const float* W    = (const float*)d_in[3];
    const int*   mask = (const int*)d_in[4];
    float* out = (float*)d_out;

    char* ws = (char*)d_ws;
    half_t* att = (half_t*)(ws);                                   // 16 MB
    half_t* Wt  = (half_t*)(ws + (size_t)16 * 1024 * 1024);        //  2 MB
    half_t* Kh  = (half_t*)(ws + (size_t)18 * 1024 * 1024);        // 16 MB
    half_t* Vt  = (half_t*)(ws + (size_t)34 * 1024 * 1024);        // 16 MB

    const size_t nelem = (size_t)B_ * T_ * D_;                     // 8M

    wconv_kernel<<<(D_ * D_) / 256, 256, 0, stream>>>(W, Wt);
    kconv_kernel<<<(unsigned)(nelem / 256), 256, 0, stream>>>(kmat, Kh);
    vconv_kernel<<<(unsigned)(nelem / 256), 256, 0, stream>>>(vmat, Vt);
    attn_kernel<<<B_ * H_ * (T_ / 128), 256, 0, stream>>>(q, Kh, Vt, mask, att);
    proj_kernel<<<(B_ * T_ / 128) * (D_ / 128), 256, 0, stream>>>(att, Wt, out);
}
